// HeteroArtistPredictor_53695681135140
// MI455X (gfx1250) — matrix operand order repaired and checked by reading, not compiled, not run on hardware
//
#include <hip/hip_runtime.h>

// ---------------------------------------------------------------------------
// Types for WMMA
// ---------------------------------------------------------------------------
typedef __attribute__((ext_vector_type(16))) _Float16 v16h;
typedef __attribute__((ext_vector_type(8)))  float    v8f;

#define ACT_NONE 0
#define ACT_RELU 1

// ---------------------------------------------------------------------------
// GEMM: C[M,N] = act(A[M,K] @ W[K,N] + bias), K = KT*32, N = NT*16.
// 256 threads = 8 waves. The whole W is converted to f16 and stored in LDS
// in the exact per-lane WMMA B-fragment layout, once per block:
//   bswz[((ks*NT + t)*32 + lane)*16 + j] = W[ks*32 + (j<8 ? (lane>>4)*8 + j : 16 + (lane>>4)*8 + j - 8)][t*16 + (lane&15)]
// so a B fragment is one contiguous 32B LDS read (2x ds_load_b128).
// Persistent grid-stride loop over 16-row M tiles amortizes the staging.
// ---------------------------------------------------------------------------
template <int KT, int NT, int ACT>
__global__ __launch_bounds__(256) void gemm_wmma(
    const float* __restrict__ A, const float* __restrict__ W,
    const float* __restrict__ bias, float* __restrict__ C, int M) {
  constexpr int K = KT * 32;
  constexpr int N = NT * 16;
  extern __shared__ _Float16 bswz[];   // K*N halfs (<= 128 KB)

  const int wave = threadIdx.x >> 5;
  const int lane = threadIdx.x & 31;
  const int m    = lane & 15;          // A row within tile / C-D column
  const int hi   = lane >> 4;

  // ---- stage W into LDS, swizzled to B-fragment order (coalesced reads) ----
  for (int i = threadIdx.x; i < K * N; i += 256) {
    const int k = i / N, col = i % N;                  // N is a power of two
    const int ks = k >> 5, kk = k & 31;
    const int l2 = (((kk >> 3) & 1) << 4) | (col & 15);      // dest lane
    const int t  = col >> 4;                           // n tile
    const int j  = (kk & 7) | ((kk >> 4) << 3);
    bswz[(((ks * NT + t) * 32 + l2) << 4) + j] = (_Float16)W[i];
  }
  __syncthreads();

  // bias per n-tile for this lane's column
  float bv[NT];
#pragma unroll
  for (int t = 0; t < NT; ++t) bv[t] = bias ? bias[t * 16 + m] : 0.0f;

  const int tiles  = (M + 15) / 16;
  const int stride = gridDim.x * 8;

  for (int rt = blockIdx.x * 8 + wave; rt < tiles; rt += stride) {
    const int row0 = rt * 16;
    const float* abase = A + (size_t)(row0 + m) * K + hi * 8;

    // prefetch next tile's A row for this lane
    __builtin_prefetch(abase + (size_t)stride * 16 * K, 0, 1);

    v8f acc[NT];
#pragma unroll
    for (int t = 0; t < NT; ++t)
#pragma unroll
      for (int v = 0; v < 8; ++v) acc[t][v] = bv[t];

#pragma unroll
    for (int ks = 0; ks < KT; ++ks) {
      // A fragment: per-lane K elements are contiguous in groups of 8 floats
      const float4 f0 = *(const float4*)(abase + ks * 32);
      const float4 f1 = *(const float4*)(abase + ks * 32 + 4);
      const float4 f2 = *(const float4*)(abase + ks * 32 + 16);
      const float4 f3 = *(const float4*)(abase + ks * 32 + 20);
      v16h a;
      a[0]  = (_Float16)f0.x; a[1]  = (_Float16)f0.y;
      a[2]  = (_Float16)f0.z; a[3]  = (_Float16)f0.w;
      a[4]  = (_Float16)f1.x; a[5]  = (_Float16)f1.y;
      a[6]  = (_Float16)f1.z; a[7]  = (_Float16)f1.w;
      a[8]  = (_Float16)f2.x; a[9]  = (_Float16)f2.y;
      a[10] = (_Float16)f2.z; a[11] = (_Float16)f2.w;
      a[12] = (_Float16)f3.x; a[13] = (_Float16)f3.y;
      a[14] = (_Float16)f3.z; a[15] = (_Float16)f3.w;

#pragma unroll
      for (int t = 0; t < NT; ++t) {
        const v16h b = *(const v16h*)(bswz + (((ks * NT + t) * 32 + lane) << 4));
        acc[t] = __builtin_amdgcn_wmma_f32_16x16x32_f16(
            false, a, false, b, (short)0, acc[t], false, false);
      }
    }

#pragma unroll
    for (int t = 0; t < NT; ++t) {
#pragma unroll
      for (int v = 0; v < 8; ++v) {
        float x = acc[t][v];
        if (ACT == ACT_RELU) x = x > 0.0f ? x : 0.0f;
        C[(size_t)(row0 + v + 8 * hi) * N + t * 16 + m] = x;
      }
    }
  }
}

// ---------------------------------------------------------------------------
// out[i] = dot(X[i, 0:128], a[0:128]) (+ bias[0]); one wave per row.
// ---------------------------------------------------------------------------
__global__ void rowdot128(const float* __restrict__ X, const float* __restrict__ a,
                          const float* __restrict__ bias, float* __restrict__ out,
                          int n) {
  const int w    = (int)((blockIdx.x * blockDim.x + threadIdx.x) >> 5);
  const int lane = threadIdx.x & 31;
  if (w >= n) return;
  const float4 x  = ((const float4*)(X + (size_t)w * 128))[lane];
  const float4 av = ((const float4*)a)[lane];
  float s = x.x * av.x + x.y * av.y + x.z * av.z + x.w * av.w;
#pragma unroll
  for (int off = 16; off > 0; off >>= 1) s += __shfl_down(s, off);
  if (lane == 0) out[w] = s + (bias ? bias[0] : 0.0f);
}

// ---------------------------------------------------------------------------
// order-preserving float <-> uint mapping for atomicMax on floats
// ---------------------------------------------------------------------------
__device__ __forceinline__ unsigned fmapu(float f) {
  unsigned u = __float_as_uint(f);
  return (u & 0x80000000u) ? ~u : (u | 0x80000000u);
}
__device__ __forceinline__ float fumap(unsigned u) {
  unsigned b = (u & 0x80000000u) ? (u & 0x7fffffffu) : ~u;
  return __uint_as_float(b);
}

__device__ __forceinline__ void edge_sd(int e, const int* es, const int* ed,
                                        int E, int& s, int& d) {
  if (e < E) { s = es[e]; d = ed[e]; } else { s = e - E; d = e - E; }
}

// pass 1: logit = leaky_relu(ls[src]+ld[dst], 0.2); elog[e]=logit; m[dst]=max
__global__ void edge_logit_max(const int* __restrict__ es, const int* __restrict__ ed,
                               int E, int L, const float* __restrict__ ls,
                               const float* __restrict__ ld, float* __restrict__ elog,
                               unsigned* __restrict__ mbuf) {
  const int e = blockIdx.x * blockDim.x + threadIdx.x;
  if (e >= E + L) return;
  int s, d; edge_sd(e, es, ed, E, s, d);
  float l = ls[s] + ld[d];
  l = l > 0.0f ? l : 0.2f * l;
  elog[e] = l;
  atomicMax(&mbuf[d], fmapu(l));
}

// pass 2: a = exp(logit - m[dst]); elog[e]=a; s[dst]+=a
__global__ void edge_expsum(const int* __restrict__ es, const int* __restrict__ ed,
                            int E, int L, const unsigned* __restrict__ mbuf,
                            float* __restrict__ elog, float* __restrict__ sbuf) {
  const int e = blockIdx.x * blockDim.x + threadIdx.x;
  if (e >= E + L) return;
  int s, d; edge_sd(e, es, ed, E, s, d);
  const float a = expf(elog[e] - fumap(mbuf[d]));
  elog[e] = a;
  atomicAdd(&sbuf[d], a);
}

// pass 3: acc[dst,:] += (a / (s[dst]+1e-16)) * hs[src,:]; one wave per edge
__global__ void edge_scatter(const int* __restrict__ es, const int* __restrict__ ed,
                             int E, int L, const float* __restrict__ elog,
                             const float* __restrict__ sbuf,
                             const float* __restrict__ hs, float* __restrict__ acc) {
  const long long idx = (long long)blockIdx.x * blockDim.x + threadIdx.x;
  const int e    = (int)(idx >> 5);
  const int lane = (int)(idx & 31);
  if (e >= E + L) return;
  int s, d; edge_sd(e, es, ed, E, s, d);
  const float coeff = elog[e] / (sbuf[d] + 1e-16f);
  const float4 h = ((const float4*)(hs + (size_t)s * 128))[lane];
  float* o = acc + (size_t)d * 128 + lane * 4;
  atomicAdd(o + 0, coeff * h.x);
  atomicAdd(o + 1, coeff * h.y);
  atomicAdd(o + 2, coeff * h.z);
  atomicAdd(o + 3, coeff * h.w);
}

// out = (elu?)(acc + b[col]), H = 128
__global__ void bias_act_ep(const float* __restrict__ acc, const float* __restrict__ b,
                            float* __restrict__ out, int n, int do_elu) {
  const long long i = (long long)blockIdx.x * blockDim.x + threadIdx.x;
  if (i >= (long long)n * 128) return;
  float v = acc[i] + b[(int)(i & 127)];
  if (do_elu) v = v > 0.0f ? v : expm1f(v);
  out[i] = v;
}

// comb[i, 0:128]=a[i,:], comb[i, 128:256]=b[i,:]
__global__ void concat2(const float* __restrict__ a, const float* __restrict__ b,
                        float* __restrict__ out, int n) {
  const long long i = (long long)blockIdx.x * blockDim.x + threadIdx.x;
  if (i >= (long long)n * 256) return;
  const int r = (int)(i >> 8), c = (int)(i & 255);
  out[i] = (c < 128) ? a[(size_t)r * 128 + c] : b[(size_t)r * 128 + (c - 128)];
}

// ---------------------------------------------------------------------------
// Host side
// ---------------------------------------------------------------------------
extern "C" void kernel_launch(void* const* d_in, const int* in_sizes, int n_in,
                              void* d_out, int out_size, void* d_ws, size_t ws_size,
                              hipStream_t stream) {
  (void)in_sizes; (void)n_in; (void)out_size; (void)ws_size;
  constexpr int NA = 20000, NW = 100000;
  constexpr int E_CR = 400000, E_IN = 600000, E_CO = 300000;

  const float* af   = (const float*)d_in[0];
  const float* wf   = (const float*)d_in[1];
  const float* a_w1 = (const float*)d_in[2];  const float* a_b1 = (const float*)d_in[3];
  const float* a_w2 = (const float*)d_in[4];  const float* a_b2 = (const float*)d_in[5];
  const float* w_w1 = (const float*)d_in[6];  const float* w_b1 = (const float*)d_in[7];
  const float* w_w2 = (const float*)d_in[8];  const float* w_b2 = (const float*)d_in[9];
  const float* cr_W = (const float*)d_in[10]; const float* cr_b = (const float*)d_in[11];
  const float* cr_as= (const float*)d_in[12]; const float* cr_ad= (const float*)d_in[13];
  const float* in_W = (const float*)d_in[14]; const float* in_b = (const float*)d_in[15];
  const float* in_as= (const float*)d_in[16]; const float* in_ad= (const float*)d_in[17];
  const float* co_W = (const float*)d_in[18]; const float* co_b = (const float*)d_in[19];
  const float* co_as= (const float*)d_in[20]; const float* co_ad= (const float*)d_in[21];
  const float* p_w1 = (const float*)d_in[22]; const float* p_b1 = (const float*)d_in[23];
  const float* p_w2 = (const float*)d_in[24]; const float* p_b2 = (const float*)d_in[25];
  const float* p_w3 = (const float*)d_in[26]; const float* p_b3 = (const float*)d_in[27];
  const int* e_cr = (const int*)d_in[28];
  const int* e_cb = (const int*)d_in[29];
  const int* e_in = (const int*)d_in[30];
  const int* e_co = (const int*)d_in[31];

  // workspace arena (floats)
  float* ws = (float*)d_ws;
  constexpr size_t F_ACC  = 0;                         // [NW,128] (also MLP tmp)
  constexpr size_t F_WXA  = F_ACC  + (size_t)NW * 128; // [NW,128]
  constexpr size_t F_WXB  = F_WXA  + (size_t)NW * 128; // [NW,128] (also hd)
  constexpr size_t F_HS   = F_WXB  + (size_t)NW * 128; // [NW,128]
  constexpr size_t F_AX   = F_HS   + (size_t)NW * 128; // [NA,128]
  constexpr size_t F_AU   = F_AX   + (size_t)NA * 128;
  constexpr size_t F_AC   = F_AU   + (size_t)NA * 128;
  constexpr size_t F_COMB = F_AC   + (size_t)NA * 128; // [NA,256]
  constexpr size_t F_H1   = F_COMB + (size_t)NA * 256; // [NA,256]
  constexpr size_t F_H2   = F_H1   + (size_t)NA * 256; // [NA,128]
  constexpr size_t F_LS   = F_H2   + (size_t)NA * 128; // [NW]
  constexpr size_t F_LD   = F_LS   + (size_t)NW;
  constexpr size_t F_MB   = F_LD   + (size_t)NW;       // uint
  constexpr size_t F_SB   = F_MB   + (size_t)NW;
  constexpr size_t F_ELOG = F_SB   + (size_t)NW;       // [700000]

  float* TMP  = ws + F_ACC;   float* ACC  = ws + F_ACC;
  float* WXA  = ws + F_WXA;   float* WXB  = ws + F_WXB;
  float* HS   = ws + F_HS;    float* HD   = ws + F_WXB;  // alias (safe)
  float* AX   = ws + F_AX;    float* AU   = ws + F_AU;   float* AC = ws + F_AC;
  float* COMB = ws + F_COMB;  float* H1   = ws + F_H1;   float* H2 = ws + F_H2;
  float* LS   = ws + F_LS;    float* LD   = ws + F_LD;
  unsigned* MB = (unsigned*)(ws + F_MB);
  float* SB   = ws + F_SB;    float* ELOG = ws + F_ELOG;

  auto gblk = [](int M) {
    unsigned b = (unsigned)(((M / 16) + 7) / 8);
    return b > 512u ? 512u : b;
  };
  auto cdiv = [](long long a, long long b) { return (unsigned)((a + b - 1) / b); };

  // LDS bytes for each GEMM shape: K*N*2
  constexpr unsigned SH_64x128  = 64  * 128 * 2;   // 16 KB
  constexpr unsigned SH_128x128 = 128 * 128 * 2;   // 32 KB
  constexpr unsigned SH_256x256 = 256 * 256 * 2;   // 128 KB
  constexpr unsigned SH_256x128 = 256 * 128 * 2;   // 64 KB

  // ---- node MLPs ----
  gemm_wmma<2, 8, ACT_RELU><<<gblk(NA), 256, SH_64x128,  stream>>>(af, a_w1, a_b1, TMP, NA);
  gemm_wmma<4, 8, ACT_NONE><<<gblk(NA), 256, SH_128x128, stream>>>(TMP, a_w2, a_b2, AX, NA);
  gemm_wmma<4, 8, ACT_RELU><<<gblk(NW), 256, SH_128x128, stream>>>(wf, w_w1, w_b1, TMP, NW);
  gemm_wmma<4, 8, ACT_NONE><<<gblk(NW), 256, SH_128x128, stream>>>(TMP, w_w2, w_b2, WXA, NW);

  // ---- generic GAT layer ----
  auto run_gat = [&](const float* xs, int n_src, const float* xd, int n_dst,
                     const int* eptr, int E, const float* W, const float* b,
                     const float* as_, const float* ad_, float* outbuf, bool elu) {
    const int L = n_src < n_dst ? n_src : n_dst;
    const int* es = eptr; const int* ed = eptr + E;
    gemm_wmma<4, 8, ACT_NONE><<<gblk(n_src), 256, SH_128x128, stream>>>(xs, W, nullptr, HS, n_src);
    const float* hd = HS;
    if (xd != nullptr) {
      gemm_wmma<4, 8, ACT_NONE><<<gblk(n_dst), 256, SH_128x128, stream>>>(xd, W, nullptr, HD, n_dst);
      hd = HD;
    }
    rowdot128<<<cdiv((long long)n_src * 32, 256), 256, 0, stream>>>(HS, as_, nullptr, LS, n_src);
    rowdot128<<<cdiv((long long)n_dst * 32, 256), 256, 0, stream>>>(hd, ad_, nullptr, LD, n_dst);
    hipMemsetAsync(MB, 0, (size_t)n_dst * 4, stream);
    hipMemsetAsync(SB, 0, (size_t)n_dst * 4, stream);
    hipMemsetAsync(ACC, 0, (size_t)n_dst * 128 * 4, stream);
    const int ET = E + L;
    edge_logit_max<<<cdiv(ET, 256), 256, 0, stream>>>(es, ed, E, L, LS, LD, ELOG, MB);
    edge_expsum<<<cdiv(ET, 256), 256, 0, stream>>>(es, ed, E, L, MB, ELOG, SB);
    edge_scatter<<<cdiv((long long)ET * 32, 256), 256, 0, stream>>>(es, ed, E, L, ELOG, SB, HS, ACC);
    bias_act_ep<<<cdiv((long long)n_dst * 128, 256), 256, 0, stream>>>(ACC, b, outbuf, n_dst, elu ? 1 : 0);
  };

  // ---- GAT stack ----
  run_gat(AX,  NA, WXA, NW, e_cr, E_CR, cr_W, cr_b, cr_as, cr_ad, WXB, true);     // creates
  run_gat(WXB, NW, nullptr, NW, e_in, E_IN, in_W, in_b, in_as, in_ad, WXA, true); // influenced
  run_gat(WXA, NW, AX, NA, e_cb, E_CR, cr_W, cr_b, cr_as, cr_ad, AU, false);      // created_by
  run_gat(AU,  NA, nullptr, NA, e_co, E_CO, co_W, co_b, co_as, co_ad, AC, true);  // collab

  // ---- prediction head ----
  concat2<<<cdiv((long long)NA * 256, 256), 256, 0, stream>>>(AU, AC, COMB, NA);
  gemm_wmma<8, 16, ACT_RELU><<<gblk(NA), 256, SH_256x256, stream>>>(COMB, p_w1, p_b1, H1, NA);
  gemm_wmma<8, 8,  ACT_RELU><<<gblk(NA), 256, SH_256x128, stream>>>(H1, p_w2, p_b2, H2, NA);
  rowdot128<<<cdiv((long long)NA * 32, 256), 256, 0, stream>>>(H2, p_w3, p_b3, (float*)d_out, NA);
}
